// MultiheadLatentAttention_69269232550603
// MI455X (gfx1250) — matrix operand order repaired and checked by reading, not compiled, not run on hardware
//
#include <hip/hip_runtime.h>
#include <hip/hip_bf16.h>

#define MLA_B 2
#define MLA_S 2048
#define MLA_H 16
// d_head = 128, d_rope = 64, qk dim = 192

typedef __attribute__((ext_vector_type(16))) __bf16 v16bf;
typedef __attribute__((ext_vector_type(8)))  float  v8f;
typedef __attribute__((ext_vector_type(4)))  unsigned int u32x4;
typedef __attribute__((ext_vector_type(4)))  int i32x4;
typedef __attribute__((ext_vector_type(8)))  int i32x8;

union BF16Frag {
  v16bf v;
  u32x4 q[2];
  unsigned short u[16];
};

__device__ inline unsigned short f2bf(float f) {
  unsigned int u = __float_as_uint(f);
  unsigned int r = u + 0x7fffu + ((u >> 16) & 1u);
  return (unsigned short)(r >> 16);
}
__device__ inline float bf2f(unsigned short h) {
  return __uint_as_float(((unsigned int)h) << 16);
}
__device__ inline v8f zero8() { v8f z = {0.f,0.f,0.f,0.f,0.f,0.f,0.f,0.f}; return z; }
__device__ inline float softplus_f(float x) {
  return (x > 20.f) ? x : __logf(1.f + __expf(x));
}

// ---------------------------------------------------------------------------
// TDM: issue an async tensor load of a [64 rows x 128 cols] bf16 tile from
// BT[N][K] (row n0, col kchunk) into LDS at lds_off.  D# per ISA 08 §8.3/8.4.
// ---------------------------------------------------------------------------
__device__ inline void tdm_issue_b(const unsigned short* BT, int n0, int K, int N,
                                   int kchunk, unsigned lds_off) {
  unsigned long long ga =
      (unsigned long long)(size_t)(BT + (size_t)n0 * K + kchunk);
  u32x4 g0;
  g0[0] = 1u;                                     // count=1, user mode
  g0[1] = lds_off;                                // lds_addr
  g0[2] = (unsigned)(ga & 0xFFFFFFFFull);         // global_addr[31:0]
  g0[3] = (unsigned)((ga >> 32) & 0x01FFFFFFull)  // global_addr[56:32]
          | 0x80000000u;                          // type=2 ("image")
  i32x8 g1;
  g1[0] = 0x00010000;                             // data_size=1 (2 bytes)
  g1[1] = (K & 0xFFFF) << 16;                     // tensor_dim0[15:0]
  g1[2] = ((K >> 16) & 0xFFFF) | ((N & 0xFFFF) << 16);   // dim0 hi | dim1 lo
  g1[3] = ((N >> 16) & 0xFFFF) | (128 << 16);     // dim1 hi | tile_dim0=128
  g1[4] = 64;                                     // tile_dim1=64, tile_dim2=0
  g1[5] = K;                                      // tensor_dim0_stride[31:0]
  g1[6] = 0;
  g1[7] = 0;
  i32x4 z4 = {0, 0, 0, 0};
#if __clang_major__ >= 23
  i32x8 z8 = {0, 0, 0, 0, 0, 0, 0, 0};
  __builtin_amdgcn_tensor_load_to_lds(g0, g1, z4, z4, z8, 0);
#else
  __builtin_amdgcn_tensor_load_to_lds(g0, g1, z4, z4, 0);
#endif
}

// ---------------------------------------------------------------------------
// elementwise converters
// ---------------------------------------------------------------------------
__global__ void cvt_f32_bf16(const float* __restrict__ in,
                             unsigned short* __restrict__ out, int n) {
  int gid = blockIdx.x * blockDim.x + threadIdx.x;
  if (gid < n) out[gid] = f2bf(in[gid]);
}

// w[K][N] (f32) -> wt[N][K] (bf16)
__global__ void transpose_cvt(const float* __restrict__ w,
                              unsigned short* __restrict__ wt, int K, int N) {
  int gid = blockIdx.x * blockDim.x + threadIdx.x;
  if (gid < K * N) {
    int k = gid / N, n = gid - k * N;
    wt[(size_t)n * K + k] = f2bf(w[gid]);
  }
}

// ---------------------------------------------------------------------------
// bf16 WMMA GEMM: C[M,N] = A[M,K] (bf16 row-major) x BT[N,K]^T
// Workgroup = 8 waves stacked in M: block tile 256x64, wave tile 32x64.
// B panel (64 x 128k) staged through LDS by the Tensor Data Mover,
// double-buffered, synchronized with s_wait_tensorcnt + workgroup barriers.
// mode 0: linear row-major output (pitch N)
// mode 1: scatter into packed qk buffer [B,H,S,192] at d=col%128
// mode 2: scatter into transposed v buffer [B,H,128,S]
// ---------------------------------------------------------------------------
__device__ inline size_t out_index(int mode, int row, int col, int N) {
  if (mode == 0) return (size_t)row * N + col;
  int b = row >> 11;            // S = 2048
  int s = row & 2047;
  int h = col >> 7;             // d_head = 128
  int d = col & 127;
  if (mode == 1) return (((size_t)(b * MLA_H + h) * MLA_S + s) * 192) + d;
  return ((size_t)(b * MLA_H + h) * 128 + d) * MLA_S + s;
}

__global__ __launch_bounds__(256)
void gemm_bf16_wmma(const unsigned short* __restrict__ A,
                    const unsigned short* __restrict__ BT,
                    float* __restrict__ Cf,
                    unsigned short* __restrict__ Cbf,
                    int M, int N, int K, int mode) {
  __shared__ __align__(16) unsigned short ldsB[2][64][128];  // 32 KB
  int wl = threadIdx.x >> 5;
  int lane = threadIdx.x & 31;
  int lrow = lane & 15, lgrp = lane >> 4;
  int nblk = N >> 6;
  int bm = blockIdx.x / nblk;
  int bn = blockIdx.x - bm * nblk;
  int m0 = bm * 256 + wl * 32;
  int n0 = bn << 6;

  unsigned lds0 = (unsigned)(size_t)&ldsB[0][0][0];
  unsigned lds1 = (unsigned)(size_t)&ldsB[1][0][0];

  int nchunks = K >> 7;   // K in chunks of 128
  if (wl == 0) tdm_issue_b(BT, n0, K, N, 0, lds0);

  v8f acc[8];
  #pragma unroll
  for (int t = 0; t < 8; t++) acc[t] = zero8();

  const unsigned short* arow0 = A + (size_t)(m0 + lrow) * K + lgrp * 8;
  const unsigned short* arow1 = arow0 + (size_t)16 * K;

  for (int c = 0; c < nchunks; c++) {
    if (wl == 0) {
      if (c + 1 < nchunks) {
        tdm_issue_b(BT, n0, K, N, (c + 1) << 7, ((c + 1) & 1) ? lds1 : lds0);
        __builtin_amdgcn_s_wait_tensorcnt((short)1);
      } else {
        __builtin_amdgcn_s_wait_tensorcnt((short)0);
      }
    }
    __syncthreads();                       // chunk c resident in ldsB[c&1]
    const unsigned short (*bufB)[128] = (const unsigned short (*)[128])ldsB[c & 1];
    #pragma unroll
    for (int kk = 0; kk < 4; kk++) {
      int kg = (c << 7) + kk * 32;
      BF16Frag a0, a1;
      a0.q[0] = *(const u32x4*)(arow0 + kg);
      a0.q[1] = *(const u32x4*)(arow0 + kg + 16);
      a1.q[0] = *(const u32x4*)(arow1 + kg);
      a1.q[1] = *(const u32x4*)(arow1 + kg + 16);
      #pragma unroll
      for (int t = 0; t < 4; t++) {
        const unsigned short* bp = &bufB[t * 16 + lrow][kk * 32 + lgrp * 8];
        BF16Frag bf;
        bf.q[0] = *(const u32x4*)(bp);
        bf.q[1] = *(const u32x4*)(bp + 16);
        acc[t]     = __builtin_amdgcn_wmma_f32_16x16x32_bf16(false, a0.v, false, bf.v, (short)0, acc[t],     false, false);
        acc[4 + t] = __builtin_amdgcn_wmma_f32_16x16x32_bf16(false, a1.v, false, bf.v, (short)0, acc[4 + t], false, false);
      }
    }
    __syncthreads();                       // all waves done with ldsB[c&1]
  }

  #pragma unroll
  for (int half = 0; half < 2; half++) {
    #pragma unroll
    for (int t = 0; t < 4; t++) {
      #pragma unroll
      for (int r = 0; r < 8; r++) {
        int row = m0 + half * 16 + r + lgrp * 8;
        int col = n0 + t * 16 + lrow;
        float val = acc[half * 4 + t][r];
        size_t oi = out_index(mode, row, col, N);
        if (Cf)  Cf[oi]  = val;
        if (Cbf) Cbf[oi] = f2bf(val);
      }
    }
  }
}

// ---------------------------------------------------------------------------
// PoPE: softplus + rotation. One thread per (b,s,h,i), i in [0,32)
// ---------------------------------------------------------------------------
__global__ void pope_q(const unsigned short* __restrict__ qr,   // [B,S,H*64] bf16
                       const float* __restrict__ delta,
                       unsigned short* __restrict__ qall) {     // [B,H,S,192] bf16
  int gid = blockIdx.x * blockDim.x + threadIdx.x;
  int i = gid & 31;
  int h = (gid >> 5) & 15;
  int bs = gid >> 9;                 // b*S + s
  int s = bs & 2047, b = bs >> 11;
  float mu1 = softplus_f(bf2f(qr[(size_t)bs * (MLA_H * 64) + h * 64 + i]));
  float mu2 = softplus_f(bf2f(qr[(size_t)bs * (MLA_H * 64) + h * 64 + 32 + i]));
  float theta = __expf(-(float)i * (9.210340371976184f / 32.0f)); // 1/10000^(2i/64)
  float ang = (float)s * theta + delta[i];
  float c = __cosf(ang), sn = __sinf(ang);
  size_t o = (((size_t)(b * MLA_H + h) * MLA_S + s) * 192) + 128 + i;
  qall[o]      = f2bf(mu1 * c - mu2 * sn);
  qall[o + 32] = f2bf(mu1 * sn + mu2 * c);
}

__global__ void pope_k(const unsigned short* __restrict__ kr,   // [B,S,64] bf16
                       const float* __restrict__ delta,
                       unsigned short* __restrict__ kall) {     // [B,H,S,192] bf16
  int gid = blockIdx.x * blockDim.x + threadIdx.x;
  int i = gid & 31;
  int h = (gid >> 5) & 15;
  int bs = gid >> 9;
  int s = bs & 2047, b = bs >> 11;
  float mu1 = softplus_f(bf2f(kr[(size_t)bs * 64 + i]));
  float mu2 = softplus_f(bf2f(kr[(size_t)bs * 64 + 32 + i]));
  float theta = __expf(-(float)i * (9.210340371976184f / 32.0f));
  float ang = (float)s * theta + delta[i];
  float c = __cosf(ang), sn = __sinf(ang);
  size_t o = (((size_t)(b * MLA_H + h) * MLA_S + s) * 192) + 128 + i;
  kall[o]      = f2bf(mu1 * c - mu2 * sn);
  kall[o + 32] = f2bf(mu1 * sn + mu2 * c);
}

// ---------------------------------------------------------------------------
// Flash attention: one wave per (b, h, 16-row q tile). Causal, online softmax.
// qall/kall: [B,H,S,192] bf16 ; vt: [B,H,128,S] bf16 ; out: [B,S,H*128] bf16
// ---------------------------------------------------------------------------
__global__ __launch_bounds__(256)
void mla_flash(const unsigned short* __restrict__ qall,
               const unsigned short* __restrict__ kall,
               const unsigned short* __restrict__ vt,
               unsigned short* __restrict__ attn_out) {
  __shared__ __align__(16) unsigned short ldsP[8][16][40]; // pitch 40 -> 80B rows
  int wave_g = (blockIdx.x * blockDim.x + threadIdx.x) >> 5;
  int wl = threadIdx.x >> 5;
  int lane = threadIdx.x & 31;
  int lrow = lane & 15, lgrp = lane >> 4;

  int qt = wave_g & 127;          // S/16 = 128 q tiles
  int bh = wave_g >> 7;           // 0..31
  int q0 = qt << 4;
  const unsigned short* qbase = qall + (size_t)bh * MLA_S * 192;
  const unsigned short* kbase = kall + (size_t)bh * MLA_S * 192;
  const unsigned short* vbase = vt   + (size_t)bh * 128 * MLA_S;

  BF16Frag qf[6];
  {
    const unsigned short* qrow = qbase + (size_t)(q0 + lrow) * 192 + lgrp * 8;
    #pragma unroll
    for (int st = 0; st < 6; st++) {
      qf[st].q[0] = *(const u32x4*)(qrow + st * 32);
      qf[st].q[1] = *(const u32x4*)(qrow + st * 32 + 16);
    }
  }

  v8f acc[8];
  #pragma unroll
  for (int t = 0; t < 8; t++) acc[t] = zero8();
  float mrun[8], lrun[8];
  #pragma unroll
  for (int r = 0; r < 8; r++) { mrun[r] = -__builtin_inff(); lrun[r] = 0.f; }

  const float scale = 0.07216878364870323f; // 1/sqrt(192)
  int ktiles = (q0 + 16 + 31) >> 5;

  for (int kt = 0; kt < ktiles; kt++) {
    int kb = kt << 5;
    v8f s0 = zero8(), s1 = zero8();
    const unsigned short* krow0 = kbase + (size_t)(kb + lrow) * 192 + lgrp * 8;
    const unsigned short* krow1 = krow0 + (size_t)16 * 192;
    #pragma unroll
    for (int st = 0; st < 6; st++) {
      BF16Frag k0, k1;
      k0.q[0] = *(const u32x4*)(krow0 + st * 32); k0.q[1] = *(const u32x4*)(krow0 + st * 32 + 16);
      k1.q[0] = *(const u32x4*)(krow1 + st * 32); k1.q[1] = *(const u32x4*)(krow1 + st * 32 + 16);
      s0 = __builtin_amdgcn_wmma_f32_16x16x32_bf16(false, qf[st].v, false, k0.v, (short)0, s0, false, false);
      s1 = __builtin_amdgcn_wmma_f32_16x16x32_bf16(false, qf[st].v, false, k1.v, (short)0, s1, false, false);
    }
    int col0 = kb + lrow, col1 = kb + 16 + lrow;
    float p0[8], p1[8], corr[8];
    #pragma unroll
    for (int r = 0; r < 8; r++) {
      int qidx = q0 + r + lgrp * 8;
      float a = s0[r] * scale; a = (col0 > qidx) ? -1e30f : a;
      float b = s1[r] * scale; b = (col1 > qidx) ? -1e30f : b;
      float mx = fmaxf(a, b);
      mx = fmaxf(mx, __shfl_xor(mx, 1, 32));
      mx = fmaxf(mx, __shfl_xor(mx, 2, 32));
      mx = fmaxf(mx, __shfl_xor(mx, 4, 32));
      mx = fmaxf(mx, __shfl_xor(mx, 8, 32));
      float nm = fmaxf(mrun[r], mx);
      corr[r] = __expf(mrun[r] - nm);
      p0[r] = __expf(a - nm);
      p1[r] = __expf(b - nm);
      lrun[r] = lrun[r] * corr[r] + p0[r] + p1[r];
      mrun[r] = nm;
    }
    #pragma unroll
    for (int t = 0; t < 8; t++)
      #pragma unroll
      for (int r = 0; r < 8; r++) acc[t][r] *= corr[r];

    #pragma unroll
    for (int r = 0; r < 8; r++) {
      int row = r + lgrp * 8;
      ldsP[wl][row][lrow]      = f2bf(p0[r]);
      ldsP[wl][row][16 + lrow] = f2bf(p1[r]);
    }
    asm volatile("" ::: "memory"); // keep LDS stores before loads (same-wave DS is in-order)
    BF16Frag pf;
    {
      const unsigned short* pr = &ldsP[wl][lrow][lgrp * 8];
      pf.q[0] = *(const u32x4*)(pr);
      pf.q[1] = *(const u32x4*)(pr + 16);
    }
    #pragma unroll
    for (int t = 0; t < 8; t++) {
      const unsigned short* vr = vbase + (size_t)(t * 16 + lrow) * MLA_S + kb + lgrp * 8;
      BF16Frag vf;
      vf.q[0] = *(const u32x4*)(vr);
      vf.q[1] = *(const u32x4*)(vr + 16);
      acc[t] = __builtin_amdgcn_wmma_f32_16x16x32_bf16(false, pf.v, false, vf.v, (short)0, acc[t], false, false);
    }
  }

  float linv[8];
  #pragma unroll
  for (int r = 0; r < 8; r++) {
    float l = lrun[r];
    l += __shfl_xor(l, 1, 32);
    l += __shfl_xor(l, 2, 32);
    l += __shfl_xor(l, 4, 32);
    l += __shfl_xor(l, 8, 32);
    linv[r] = 1.0f / l;
  }
  int b = bh >> 4, h = bh & 15;
  #pragma unroll
  for (int t = 0; t < 8; t++) {
    #pragma unroll
    for (int r = 0; r < 8; r++) {
      int row = q0 + r + lgrp * 8;
      int col = h * 128 + t * 16 + lrow;
      attn_out[(size_t)(b * MLA_S + row) * (MLA_H * 128) + col] = f2bf(acc[t][r] * linv[r]);
    }
  }
}

// ---------------------------------------------------------------------------
// host-side orchestration
// ---------------------------------------------------------------------------
extern "C" void kernel_launch(void* const* d_in, const int* in_sizes, int n_in,
                              void* d_out, int out_size, void* d_ws, size_t ws_size,
                              hipStream_t stream) {
  (void)in_sizes; (void)n_in; (void)out_size; (void)ws_size;
  const float* x     = (const float*)d_in[0];
  const float* w_dkv = (const float*)d_in[1];
  const float* w_uk  = (const float*)d_in[2];
  const float* w_uv  = (const float*)d_in[3];
  const float* w_dq  = (const float*)d_in[4];
  const float* w_uq  = (const float*)d_in[5];
  const float* w_qr  = (const float*)d_in[6];
  const float* w_kr  = (const float*)d_in[7];
  const float* w_out = (const float*)d_in[8];
  const float* delta = (const float*)d_in[9];

  const size_t M = (size_t)MLA_B * MLA_S;  // 4096
  char* ws = (char*)d_ws;
  size_t off = 0;
  auto alloc = [&](size_t elems) -> unsigned short* {
    unsigned short* p = (unsigned short*)(ws + off);
    off += (elems * 2 + 255) & ~(size_t)255;
    return p;
  };

  unsigned short* xbf    = alloc(M * 2048);
  unsigned short* wt_dkv = alloc((size_t)512 * 2048);
  unsigned short* wt_uk  = alloc((size_t)2048 * 512);
  unsigned short* wt_uv  = alloc((size_t)2048 * 512);
  unsigned short* wt_dq  = alloc((size_t)1024 * 2048);
  unsigned short* wt_uq  = alloc((size_t)2048 * 1024);
  unsigned short* wt_qr  = alloc((size_t)1024 * 1024);
  unsigned short* wt_kr  = alloc((size_t)64 * 2048);
  unsigned short* wt_out = alloc((size_t)2048 * 2048);
  unsigned short* ckv    = alloc(M * 512);
  unsigned short* cq     = alloc(M * 1024);
  unsigned short* qrpre  = alloc(M * 1024);
  unsigned short* krpre  = alloc(M * 64);
  unsigned short* qall   = alloc((size_t)MLA_B * MLA_H * MLA_S * 192);
  unsigned short* kall   = alloc((size_t)MLA_B * MLA_H * MLA_S * 192);
  unsigned short* vtb    = alloc((size_t)MLA_B * MLA_H * 128 * MLA_S);
  unsigned short* attn   = alloc(M * 2048);

  auto ew_grid = [](size_t n) { return dim3((unsigned)((n + 255) / 256)); };

  cvt_f32_bf16<<<ew_grid(M * 2048), 256, 0, stream>>>(x, xbf, (int)(M * 2048));
  transpose_cvt<<<ew_grid((size_t)2048 * 512),  256, 0, stream>>>(w_dkv, wt_dkv, 2048, 512);
  transpose_cvt<<<ew_grid((size_t)512 * 2048),  256, 0, stream>>>(w_uk,  wt_uk,  512, 2048);
  transpose_cvt<<<ew_grid((size_t)512 * 2048),  256, 0, stream>>>(w_uv,  wt_uv,  512, 2048);
  transpose_cvt<<<ew_grid((size_t)2048 * 1024), 256, 0, stream>>>(w_dq,  wt_dq,  2048, 1024);
  transpose_cvt<<<ew_grid((size_t)1024 * 2048), 256, 0, stream>>>(w_uq,  wt_uq,  1024, 2048);
  transpose_cvt<<<ew_grid((size_t)1024 * 1024), 256, 0, stream>>>(w_qr,  wt_qr,  1024, 1024);
  transpose_cvt<<<ew_grid((size_t)2048 * 64),   256, 0, stream>>>(w_kr,  wt_kr,  2048, 64);
  transpose_cvt<<<ew_grid((size_t)2048 * 2048), 256, 0, stream>>>(w_out, wt_out, 2048, 2048);

  auto launch_gemm = [&](const unsigned short* A, const unsigned short* BT,
                         float* Cf, unsigned short* Cbf, int Mm, int Nn, int Kk, int mode) {
    int blocks = (Mm / 256) * (Nn / 64);
    gemm_bf16_wmma<<<dim3(blocks), dim3(256), 0, stream>>>(A, BT, Cf, Cbf, Mm, Nn, Kk, mode);
  };

  launch_gemm(xbf, wt_dkv, nullptr, ckv,  (int)M,  512, 2048, 0);  // c_kv
  launch_gemm(ckv, wt_uk,  nullptr, kall, (int)M, 2048,  512, 1);  // key_c -> kall[...,0:128]
  launch_gemm(ckv, wt_uv,  nullptr, vtb,  (int)M, 2048,  512, 2);  // value -> v^T
  launch_gemm(xbf, wt_dq,  nullptr, cq,   (int)M, 1024, 2048, 0);  // c_q
  launch_gemm(cq,  wt_uq,  nullptr, qall, (int)M, 2048, 1024, 1);  // query_c -> qall[...,0:128]
  launch_gemm(cq,  wt_qr,  nullptr, qrpre,(int)M, 1024, 1024, 0);  // rope pre (q)
  launch_gemm(xbf, wt_kr,  nullptr, krpre,(int)M,   64, 2048, 0);  // rope pre (k)

  size_t nrope = (size_t)MLA_B * MLA_S * MLA_H * 32;
  pope_q<<<ew_grid(nrope), 256, 0, stream>>>(qrpre, delta, qall);
  pope_k<<<ew_grid(nrope), 256, 0, stream>>>(krpre, delta, kall);

  mla_flash<<<dim3(512), dim3(256), 0, stream>>>(qall, kall, vtb, attn);

  launch_gemm(attn, wt_out, (float*)d_out, nullptr, (int)M, 2048, 2048, 0);
}
